// GraphAttentionLayer_36550171689279
// MI455X (gfx1250) — matrix operand order repaired and checked by reading, not compiled, not run on hardware
//
#include <hip/hip_runtime.h>

#define ALPHA_SLOPE 0.2f
#define NEG_INF_VAL -9.0e15f

typedef __bf16 bf16;
typedef __bf16 v16bf __attribute__((ext_vector_type(16)));
typedef float  v8f   __attribute__((ext_vector_type(8)));

struct B32x2 { uint4 lo, hi; };

// Assemble a 32-byte WMMA fragment from two 16-byte LDS chunks (-> 2x ds_load_b128)
__device__ __forceinline__ v16bf frag2(const bf16* p0, const bf16* p1) {
    B32x2 t;
    t.lo = *(const uint4*)p0;
    t.hi = *(const uint4*)p1;
    return __builtin_bit_cast(v16bf, t);
}

// Pre-bake: W (f32, [v][u]) -> transposed bf16 [u][v] (B-fragment friendly: lane=column u,
// K=v contiguous), and Asum mask. Both tiny, L2-resident for the main kernel.
__global__ __launch_bounds__(256) void gat_prep(const float* __restrict__ A,
                                                const float* __restrict__ W,
                                                bf16* __restrict__ WbfT,
                                                float* __restrict__ maskG) {
    const int tid = threadIdx.x;
    for (int i = tid; i < 4096; i += 256) {
        const int v = i >> 6, u = i & 63;
        WbfT[u * 64 + v] = (bf16)W[i];                       // W[v][u] -> WbfT[u][v]
        maskG[i] = A[i] + A[4096 + i] + A[8192 + i];        // Asum[i][j]
    }
}

__global__ __launch_bounds__(256) void gat_main(const float* __restrict__ x,
                                                const float* __restrict__ a,
                                                const bf16*  __restrict__ WbfT,
                                                const float* __restrict__ maskG,
                                                float* __restrict__ out) {
    __shared__ __align__(16) bf16  Xbf[64 * 64];    // X[c][v] (A of GEMM1)
    __shared__ __align__(16) bf16  Wsh[64 * 64];    // W^T[u][v] (B of GEMM1)
    __shared__ __align__(16) bf16  Ybf[64 * 64];    // Y[c][u] (A of GEMM2)
    __shared__ __align__(16) bf16  attb[64 * 64];   // att[v][j] (B of GEMM2)
    __shared__ __align__(16) float masksh[64 * 64];
    __shared__ float fsh[64], gsh[64], a1s[64], a2s[64];

    const int tid   = threadIdx.x;
    const int lane  = tid & 31;
    const int wave  = tid >> 5;
    // Tile assignment: wave handles (mt, nt0) and (mt, nt1) -> shared A-fragment.
    const int mt    = wave >> 1;
    const int nt0   = (wave & 1) * 2;
    const int nt1   = nt0 + 1;
    const int mrow  = lane & 15;   // A row / B column / D column within tile
    const int khalf = lane >> 4;   // which K-half this lane holds

    // Per-lane staging coordinates for the x tile (row c, 16 floats from vb).
    const int cRow = tid >> 2;
    const int vb   = (tid & 3) << 4;

    // Per-WGP-resident tiles: fill once per block.
    {
        const uint4* ps = (const uint4*)WbfT;
        uint4* pd = (uint4*)Wsh;
        pd[tid]       = ps[tid];
        pd[tid + 256] = ps[tid + 256];
        const float4* ms = (const float4*)maskG;
        float4* md = (float4*)masksh;
        #pragma unroll
        for (int q = 0; q < 4; ++q) md[tid + q * 256] = ms[tid + q * 256];
        if (tid < 64) { a1s[tid] = a[tid]; a2s[tid] = a[64 + tid]; }
    }

    // Each block processes 4 consecutive (n,t) slices.
    const int b0 = blockIdx.x * 4;
    for (int it = 0; it < 4; ++it) {
        const int b    = b0 + it;
        const int nIdx = b >> 9;     // T = 512
        const int tIdx = b & 511;

        // ---- Stage X[c][v] (64x64 f32) -> bf16 LDS; reset f/g accumulators ----
        {
            const float4* p =
                (const float4*)(x + (((size_t)(nIdx * 64 + cRow) * 512 + tIdx) * 64 + vb));
            bf16* dst = &Xbf[cRow * 64 + vb];
            #pragma unroll
            for (int q = 0; q < 4; ++q) {
                float4 r = p[q];
                dst[q * 4 + 0] = (bf16)r.x; dst[q * 4 + 1] = (bf16)r.y;
                dst[q * 4 + 2] = (bf16)r.z; dst[q * 4 + 3] = (bf16)r.w;
            }
            if (tid < 64) { fsh[tid] = 0.f; gsh[tid] = 0.f; }
            // Pull next slice's tile toward L2/L0 while this slice computes
            // (global_prefetch_b8: one cacheline per lane, no counter cost).
            if (it < 3) {
                const int bn = b + 1;
                const float* np =
                    x + (((size_t)((bn >> 9) * 64 + cRow) * 512 + (bn & 511)) * 64 + vb);
                __builtin_prefetch(np, 0, 3);
            }
        }
        __syncthreads();

        // ---- GEMM1: Y[c][u] = sum_v X[c][v] * W[v][u] ----
        v8f acc0 = {}; v8f acc1 = {};
        #pragma unroll
        for (int kh = 0; kh < 2; ++kh) {
            const int ka = kh * 32 + khalf * 16;   // A: K contiguous 16 per lane
            const bf16* ap = &Xbf[(mt * 16 + mrow) * 64 + ka];
            v16bf af = frag2(ap, ap + 8);
            const int kb = kh * 32 + khalf * 16;   // B: K contiguous 16 per lane
            const bf16* bp0 = &Wsh[(nt0 * 16 + mrow) * 64 + kb];
            const bf16* bp1 = &Wsh[(nt1 * 16 + mrow) * 64 + kb];
            v16bf bf0 = frag2(bp0, bp0 + 8);
            v16bf bf1 = frag2(bp1, bp1 + 8);
            acc0 = __builtin_amdgcn_wmma_f32_16x16x32_bf16(false, af, false, bf0,
                                                           (short)0, acc0, false, false);
            acc1 = __builtin_amdgcn_wmma_f32_16x16x32_bf16(false, af, false, bf1,
                                                           (short)0, acc1, false, false);
        }

        // Spill Y (bf16) + fold f/g partial dots out of the f32 accumulators.
        {
            float pf0 = 0.f, pg0 = 0.f, pf1 = 0.f, pg1 = 0.f;
            #pragma unroll
            for (int r = 0; r < 8; ++r) {
                const int c = mt * 16 + khalf * 8 + r;  // D row = c
                const float a1v = a1s[c], a2v = a2s[c];
                const float y0 = acc0[r], y1 = acc1[r];
                Ybf[c * 64 + nt0 * 16 + mrow] = (bf16)y0;
                Ybf[c * 64 + nt1 * 16 + mrow] = (bf16)y1;
                pf0 += y0 * a1v; pg0 += y0 * a2v;
                pf1 += y1 * a1v; pg1 += y1 * a2v;
            }
            atomicAdd(&fsh[nt0 * 16 + mrow], pf0);   // ds_add_f32
            atomicAdd(&gsh[nt0 * 16 + mrow], pg0);
            atomicAdd(&fsh[nt1 * 16 + mrow], pf1);
            atomicAdd(&gsh[nt1 * 16 + mrow], pg1);
        }
        __syncthreads();

        // ---- Masked leaky-relu scores + row softmax (f32), 4 lanes per row ----
        {
            const int row = tid >> 2;
            const int q   = tid & 3;
            const float fi = fsh[row];
            float ev[16];
            float mx = -3.0e38f;
            #pragma unroll
            for (int jj = 0; jj < 16; ++jj) {
                const int j = q * 16 + jj;
                const float s  = fi + gsh[j];
                const float lr = s > 0.f ? s : ALPHA_SLOPE * s;
                const float vv = masksh[row * 64 + j] > 0.f ? lr : NEG_INF_VAL;
                ev[jj] = vv;
                mx = fmaxf(mx, vv);
            }
            mx = fmaxf(mx, __shfl_xor(mx, 1, 4));
            mx = fmaxf(mx, __shfl_xor(mx, 2, 4));
            float sum = 0.f;
            #pragma unroll
            for (int jj = 0; jj < 16; ++jj) {
                const float e = __expf(ev[jj] - mx);
                ev[jj] = e;
                sum += e;
            }
            sum += __shfl_xor(sum, 1, 4);
            sum += __shfl_xor(sum, 2, 4);
            const float inv = 1.0f / sum;
            #pragma unroll
            for (int jj = 0; jj < 16; ++jj)
                attb[row * 64 + q * 16 + jj] = (bf16)(ev[jj] * inv);
        }
        __syncthreads();

        // ---- GEMM2: out[c][v] = sum_j Y[c][j] * att[v][j] (att rows ARE B columns) ----
        v8f o0 = {}; v8f o1 = {};
        #pragma unroll
        for (int kh = 0; kh < 2; ++kh) {
            const int ka = kh * 32 + khalf * 16;
            const bf16* ap = &Ybf[(mt * 16 + mrow) * 64 + ka];
            v16bf af = frag2(ap, ap + 8);
            const int kb = kh * 32 + khalf * 16;
            const bf16* bp0 = &attb[(nt0 * 16 + mrow) * 64 + kb];
            const bf16* bp1 = &attb[(nt1 * 16 + mrow) * 64 + kb];
            v16bf bf0 = frag2(bp0, bp0 + 8);
            v16bf bf1 = frag2(bp1, bp1 + 8);
            o0 = __builtin_amdgcn_wmma_f32_16x16x32_bf16(false, af, false, bf0,
                                                         (short)0, o0, false, false);
            o1 = __builtin_amdgcn_wmma_f32_16x16x32_bf16(false, af, false, bf1,
                                                         (short)0, o1, false, false);
        }

        // ---- Scatter to out[n][c][t][v] (16 consecutive lanes -> 64B runs) ----
        #pragma unroll
        for (int r = 0; r < 8; ++r) {
            const int c = mt * 16 + khalf * 8 + r;
            const size_t base = ((size_t)(nIdx * 64 + c) * 512 + tIdx) * 64;
            out[base + nt0 * 16 + mrow] = o0[r];
            out[base + nt1 * 16 + mrow] = o1[r];
        }
        __syncthreads();   // Xbf / fsh reuse next iteration
    }
}

extern "C" void kernel_launch(void* const* d_in, const int* in_sizes, int n_in,
                              void* d_out, int out_size, void* d_ws, size_t ws_size,
                              hipStream_t stream) {
    (void)in_sizes; (void)n_in; (void)out_size; (void)ws_size;
    const float* x = (const float*)d_in[0];
    const float* A = (const float*)d_in[1];
    const float* W = (const float*)d_in[2];
    const float* a = (const float*)d_in[3];
    float* out = (float*)d_out;

    bf16*  WbfT  = (bf16*)d_ws;                          // 8 KB
    float* maskG = (float*)((char*)d_ws + 64 * 64 * 2);  // 16 KB, 16B-aligned

    gat_prep<<<1, 256, 0, stream>>>(A, W, WbfT, maskG);
    // 16*512 = 8192 (n,t) slices, 4 per block
    gat_main<<<8192 / 4, 256, 0, stream>>>(x, a, WbfT, maskG, out);
}
